// DeformableAlign_10471130267859
// MI455X (gfx1250) — matrix operand order repaired and checked by reading, not compiled, not run on hardware
//
#include <hip/hip_runtime.h>
#include <hip/hip_bf16.h>
#include <math.h>

typedef __attribute__((ext_vector_type(16))) _Float16 v16h;
typedef __attribute__((ext_vector_type(8)))  _Float16 v8h;
typedef __attribute__((ext_vector_type(8)))  float    v8f;

#define BATCH 8
#define CCH   256
#define HH    128
#define WW    128
#define C2    512
#define HID   64
#define HWSZ  (HH*WW)                 // 16384
#define NCHW  ((size_t)BATCH*CCH*HWSZ) // 33554432

// ---------------------------------------------------------------------------
// Kernel 1: repack p1_w (64x512 f32) into f16 WMMA B-fragment order, and fold
// p2_w (18x64) / p2_b into 4 reduced offset vectors qw[4][64], qb[4].
// B-frag layout (32x16 KxN f16, wave32): lane L -> n = L%16, holds 16 halves (contig in buffer):
// K = (L/16)*8 + i (i<8), 16 + (L/16)*8 + i-8 (i>=8). Fragment f = kslice*4 + ntile, each 512 halves.
// ---------------------------------------------------------------------------
__global__ void da_pack_kernel(const float* __restrict__ p1_w,
                               const float* __restrict__ p2_w,
                               const float* __restrict__ p2_b,
                               _Float16* __restrict__ p1h,
                               float* __restrict__ qw,
                               float* __restrict__ qb) {
    const int tid = threadIdx.x;
    for (int o = tid; o < HID * C2; o += 256) {
        int f = o >> 9, within = o & 511;
        int s = f >> 2, t = f & 3;
        int L = within >> 4, i = within & 15;
        int n = t * 16 + (L & 15);
        int k = s * 32 + ((i & 8) << 1) + (L >> 4) * 8 + (i & 7);
        p1h[o] = (_Float16)p1_w[n * C2 + k];
    }
    // qw[j][n]: j=0 rgb-x (rows 0,2,4,6), j=1 rgb-y (1,3,5,7),
    //           j=2 tir-x (8,10,12,14),  j=3 tir-y (9,11,13,15)
    {
        int j = tid >> 6, n = tid & 63;
        int base = (j >> 1) * 8 + (j & 1);
        float s = 0.25f * (p2_w[(base + 0) * 64 + n] + p2_w[(base + 2) * 64 + n] +
                           p2_w[(base + 4) * 64 + n] + p2_w[(base + 6) * 64 + n]);
        qw[j * 64 + n] = s;
    }
    if (tid < 4) {
        int base = (tid >> 1) * 8 + (tid & 1);
        qb[tid] = 0.25f * (p2_b[base] + p2_b[base + 2] + p2_b[base + 4] + p2_b[base + 6]);
    }
}

// ---------------------------------------------------------------------------
// Kernel 2: pooled[b][c] = mean over HxW of concat channel c (c<256 rgb).
// ---------------------------------------------------------------------------
__global__ void da_pool_kernel(const float* __restrict__ rgb,
                               const float* __restrict__ tir,
                               float* __restrict__ pooled) {
    const int bc = blockIdx.x;
    const int b = bc >> 9, c = bc & 511;
    const float* src = ((c < CCH) ? rgb : tir) +
                       ((size_t)(b * CCH + (c & (CCH - 1)))) * HWSZ;
    float s = 0.f;
    for (int i = threadIdx.x; i < HWSZ; i += 256) s += src[i];
    __shared__ float red[256];
    red[threadIdx.x] = s;
    __syncthreads();
    for (int st = 128; st > 0; st >>= 1) {
        if (threadIdx.x < st) red[threadIdx.x] += red[threadIdx.x + st];
        __syncthreads();
    }
    if (threadIdx.x == 0) pooled[bc] = red[0] * (1.f / (float)HWSZ);
}

// ---------------------------------------------------------------------------
// Kernel 3: modulation MLP: relu(pooled@m1^T+b) -> sigmoid(@m2^T+b). 8 blocks.
// ---------------------------------------------------------------------------
__global__ void da_mod_kernel(const float* __restrict__ pooled,
                              const float* __restrict__ m1_w,
                              const float* __restrict__ m1_b,
                              const float* __restrict__ m2_w,
                              const float* __restrict__ m2_b,
                              float* __restrict__ modv) {
    const int b = blockIdx.x, tid = threadIdx.x;
    __shared__ float hid[HID];
    if (tid < HID) {
        float s = m1_b[tid];
        const float* p = pooled + b * C2;
        const float* w = m1_w + tid * C2;
        for (int k = 0; k < C2; ++k) s += p[k] * w[k];
        hid[tid] = fmaxf(s, 0.f);
    }
    __syncthreads();
    float s = m2_b[tid];
    const float* w2 = m2_w + tid * HID;
    for (int j = 0; j < HID; ++j) s += hid[j] * w2[j];
    modv[b * CCH + tid] = 1.f / (1.f + __expf(-s));
}

// ---------------------------------------------------------------------------
// Kernel 4: fused depthwise 3x3 -> WMMA GEMM (512->64, f16 in / f32 acc) ->
// ReLU -> reduced offset head -> bilinear border sampling + modulation ->
// quality. One 32-pixel row segment per 128-thread (4 wave32) workgroup.
// ---------------------------------------------------------------------------
#define ASTRIDE 520   // 512 + 8 halves pad (bank spread for ds_read_b128)
#define HSTRIDE 65    // 64 + 1 f32 pad

__global__ void __launch_bounds__(128)
da_main_kernel(const float* __restrict__ rgb,
               const float* __restrict__ tir,
               const float* __restrict__ dw_w,
               const float* __restrict__ dw_b,
               const float* __restrict__ p1_b,
               const _Float16* __restrict__ p1h,
               const float* __restrict__ qw,
               const float* __restrict__ qb,
               const float* __restrict__ modv,
               float* __restrict__ out) {
    __shared__ _Float16 Astage[32 * ASTRIDE];   // 32 px x 512 ch dwconv out (f16)
    __shared__ float    hid[32 * HSTRIDE];      // 32 px x 64 hidden (post-ReLU)
    __shared__ float    offs[4 * 32];           // [j][px] offsets
    __shared__ float    qpart[4 * 32];          // per-wave |diff| partials

    const int tid  = threadIdx.x;
    const int lane = tid & 31;
    const int wv   = tid >> 5;
    const int b  = blockIdx.z;
    const int h  = blockIdx.y;
    const int w0 = blockIdx.x * 32;
    const int xc = w0 + lane;

    // ---- Phase 1: depthwise 3x3 (zero-pad SAME), pixel-per-lane, 128 ch/wave
    {
        const bool xm_ok = (xc > 0), xp_ok = (xc < WW - 1);
        const int c0 = wv * 128;
#pragma unroll 1
        for (int c = c0; c < c0 + 128; ++c) {
            const float* src = ((c < CCH) ? rgb : tir) +
                               ((size_t)(b * CCH + (c & (CCH - 1)))) * HWSZ;
            const float* wp = dw_w + c * 9;
            float acc = dw_b[c];
#pragma unroll
            for (int ky = 0; ky < 3; ++ky) {
                const int y = h + ky - 1;
                if (y >= 0 && y < HH) {
                    const float* row = src + y * WW + xc;
                    acc += wp[ky * 3 + 1] * row[0];
                    if (xm_ok) acc += wp[ky * 3 + 0] * row[-1];
                    if (xp_ok) acc += wp[ky * 3 + 2] * row[1];
                }
            }
            Astage[lane * ASTRIDE + c] = (_Float16)acc;
        }
    }
    __syncthreads();

    // ---- Phase 2: GEMM hid[32x64] = dwout[32x512] @ p1_w^T via WMMA f16
    {
        const int mt  = wv >> 1;          // M tile (16 px)
        const int ntb = (wv & 1) * 2;     // first of two N tiles
        const int lm = lane & 15, lh = lane >> 4;
        const _Float16* Ab = &Astage[(mt * 16 + lm) * ASTRIDE + lh * 8];
        v8f acc0 = {}; v8f acc1 = {};
#pragma unroll 4
        for (int s = 0; s < 16; ++s) {
            // A fragment: 16-bit A 16x32 layout (K halves 0..7 then 16..23 per half-wave)
            v8h alo = *(const v8h*)(Ab + s * 32);
            v8h ahi = *(const v8h*)(Ab + s * 32 + 16);
            v16h a = __builtin_shufflevector(alo, ahi,
                     0,1,2,3,4,5,6,7,8,9,10,11,12,13,14,15);
            // B fragments: prepacked, lane-contiguous 16 halves
            const _Float16* bp0 = p1h + ((s * 4 + ntb) * 512) + lane * 16;
            v8h b0lo = *(const v8h*)(bp0);
            v8h b0hi = *(const v8h*)(bp0 + 8);
            v16h b0 = __builtin_shufflevector(b0lo, b0hi,
                      0,1,2,3,4,5,6,7,8,9,10,11,12,13,14,15);
            const _Float16* bp1 = bp0 + 512;
            v8h b1lo = *(const v8h*)(bp1);
            v8h b1hi = *(const v8h*)(bp1 + 8);
            v16h b1 = __builtin_shufflevector(b1lo, b1hi,
                      0,1,2,3,4,5,6,7,8,9,10,11,12,13,14,15);
            acc0 = __builtin_amdgcn_wmma_f32_16x16x32_f16(
                       false, a, false, b0, (short)0, acc0, false, false);
            acc1 = __builtin_amdgcn_wmma_f32_16x16x32_f16(
                       false, a, false, b1, (short)0, acc1, false, false);
        }
        // C/D layout: VGPR r -> M = r + (lane/16)*8, N = lane%16
        const int n0 = ntb * 16 + lm;
#pragma unroll
        for (int r = 0; r < 8; ++r) {
            const int px = mt * 16 + lh * 8 + r;
            hid[px * HSTRIDE + n0]      = fmaxf(acc0[r] + p1_b[n0], 0.f);
            hid[px * HSTRIDE + n0 + 16] = fmaxf(acc1[r] + p1_b[n0 + 16], 0.f);
        }
    }
    __syncthreads();

    // ---- Phase 2b: reduced offset head: offs[j][px] = qb[j] + hid[px].qw[j]
    {
        const int j = wv, px = lane;
        float s = qb[j];
        const float* qwj = qw + j * 64;
        const float* hp = &hid[px * HSTRIDE];
#pragma unroll 8
        for (int n = 0; n < HID; ++n) s += hp[n] * qwj[n];
        offs[j * 32 + px] = s;
    }
    __syncthreads();

    // ---- Phase 3: bilinear (border, align_corners) sample + mod + quality.
    // ix = clamp(w + 0.1*0.5*(W-1)*off, 0, W-1) = clamp(w + 6.35*off, 0, 127)
    float w00r, w01r, w10r, w11r, w00t, w01t, w10t, w11t;
    int i00r, i01r, i10r, i11r, i00t, i01t, i10t, i11t;
    {
        const float oxr = offs[0 * 32 + lane], oyr = offs[1 * 32 + lane];
        const float oxt = offs[2 * 32 + lane], oyt = offs[3 * 32 + lane];
        float ix = fminf(fmaxf((float)xc + 6.35f * oxr, 0.f), 127.f);
        float iy = fminf(fmaxf((float)h  + 6.35f * oyr, 0.f), 127.f);
        float x0 = floorf(ix), y0 = floorf(iy);
        int x0i = (int)x0, y0i = (int)y0;
        int x1i = (x0i + 1 < WW) ? x0i + 1 : WW - 1;
        int y1i = (y0i + 1 < HH) ? y0i + 1 : HH - 1;
        float wx = ix - x0, wy = iy - y0;
        i00r = y0i * WW + x0i; i01r = y0i * WW + x1i;
        i10r = y1i * WW + x0i; i11r = y1i * WW + x1i;
        w00r = (1.f - wx) * (1.f - wy); w01r = wx * (1.f - wy);
        w10r = (1.f - wx) * wy;         w11r = wx * wy;

        ix = fminf(fmaxf((float)xc + 6.35f * oxt, 0.f), 127.f);
        iy = fminf(fmaxf((float)h  + 6.35f * oyt, 0.f), 127.f);
        x0 = floorf(ix); y0 = floorf(iy);
        x0i = (int)x0; y0i = (int)y0;
        x1i = (x0i + 1 < WW) ? x0i + 1 : WW - 1;
        y1i = (y0i + 1 < HH) ? y0i + 1 : HH - 1;
        wx = ix - x0; wy = iy - y0;
        i00t = y0i * WW + x0i; i01t = y0i * WW + x1i;
        i10t = y1i * WW + x0i; i11t = y1i * WW + x1i;
        w00t = (1.f - wx) * (1.f - wy); w01t = wx * (1.f - wy);
        w10t = (1.f - wx) * wy;         w11t = wx * wy;
    }

    float* out_rgb = out;
    float* out_tir = out + NCHW;
    float* out_q   = out + 2 * NCHW;
    const size_t spatial = (size_t)h * WW + xc;
    const int cb = wv * 64;
    float qacc = 0.f;
#pragma unroll 2
    for (int cc = 0; cc < 64; ++cc) {
        const int c = cb + cc;
        const size_t fo = ((size_t)(b * CCH + c)) * HWSZ;
        const float* R = rgb + fo;
        const float* T = tir + fo;
        const float mc = modv[b * CCH + c];
        float rs = (R[i00r] * w00r + R[i01r] * w01r +
                    R[i10r] * w10r + R[i11r] * w11r) * mc;
        float ts = (T[i00t] * w00t + T[i01t] * w01t +
                    T[i10t] * w10t + T[i11t] * w11t) * mc;
        out_rgb[fo + spatial] = rs;
        out_tir[fo + spatial] = ts;
        qacc += fabsf(rs - ts);
    }
    qpart[wv * 32 + lane] = qacc;
    __syncthreads();
    if (tid < 32) {
        float q = qpart[tid] + qpart[32 + tid] + qpart[64 + tid] + qpart[96 + tid];
        q = 1.f - q * (1.f / (float)CCH);
        out_q[(size_t)b * HWSZ + spatial] = 1.f / (1.f + __expf(-q));
    }
}

// ---------------------------------------------------------------------------
extern "C" void kernel_launch(void* const* d_in, const int* in_sizes, int n_in,
                              void* d_out, int out_size, void* d_ws, size_t ws_size,
                              hipStream_t stream) {
    (void)in_sizes; (void)n_in; (void)out_size; (void)ws_size;
    const float* rgb  = (const float*)d_in[0];
    const float* tir  = (const float*)d_in[1];
    const float* dw_w = (const float*)d_in[2];
    const float* dw_b = (const float*)d_in[3];
    const float* p1_w = (const float*)d_in[4];
    const float* p1_b = (const float*)d_in[5];
    const float* p2_w = (const float*)d_in[6];
    const float* p2_b = (const float*)d_in[7];
    const float* m1_w = (const float*)d_in[8];
    const float* m1_b = (const float*)d_in[9];
    const float* m2_w = (const float*)d_in[10];
    const float* m2_b = (const float*)d_in[11];
    float* out = (float*)d_out;

    // workspace layout
    char* ws = (char*)d_ws;
    _Float16* p1h   = (_Float16*)(ws + 0);       // 65536 B
    float*    qw    = (float*)(ws + 65536);      // 1024 B
    float*    qb    = (float*)(ws + 66560);      // 16 B (pad to 64)
    float*    pooled= (float*)(ws + 66624);      // 16384 B
    float*    modv  = (float*)(ws + 83008);      // 8192 B

    da_pack_kernel<<<1, 256, 0, stream>>>(p1_w, p2_w, p2_b, p1h, qw, qb);
    da_pool_kernel<<<BATCH * C2, 256, 0, stream>>>(rgb, tir, pooled);
    da_mod_kernel<<<BATCH, 256, 0, stream>>>(pooled, m1_w, m1_b, m2_w, m2_b, modv);
    da_main_kernel<<<dim3(WW / 32, HH, BATCH), 128, 0, stream>>>(
        rgb, tir, dw_w, dw_b, p1_b, p1h, qw, qb, modv, out);
}
